// DecoderPrediction_47485158425196
// MI455X (gfx1250) — matrix operand order repaired and checked by reading, not compiled, not run on hardware
//
#include <hip/hip_runtime.h>
#include <stdint.h>

// ---------------- problem constants (match reference) ----------------
#define B_    128
#define T_    64
#define DENC  768
#define UU    256      // LSTM units
#define EE    256      // label embedding dim
#define VV    48       // label vocab
#define N4    1024     // 4*UU gate width
#define KTOT  1280     // EE + UU + DENC  (K order: emb | h | hard)
#define KT_N  40       // KTOT / 32
#define ROWS  16       // batch rows per workgroup (one WMMA M-tile)
#define XSTR  1288     // padded x row stride (bf16): 2576B = 644 dw, 644%64=4 -> bank spread

typedef __attribute__((ext_vector_type(16))) __bf16        v16bf;
typedef __attribute__((ext_vector_type(8)))  __bf16        v8bf;
typedef __attribute__((ext_vector_type(8)))  float         v8f;
typedef __attribute__((ext_vector_type(4))) unsigned int   u32x4;
typedef __attribute__((ext_vector_type(8)))  int           i32x8;
typedef __attribute__((ext_vector_type(4)))  int           i32x4;

// =====================================================================
// Prep 1: pack [W;Uw] (f32) into bf16 WMMA-B-operand order.
// Wpk layout: [tileN(64)][kt(40)][lane(32)][j(16)]  (bf16)
//   lane L: col = tileN*16 + (L&15), kbase = kt*32 + (L>>4)*16
// K-order: k<256 -> W[k][col]; 256<=k<512 -> Uw[k-256][col]; k>=512 -> W[256+(k-512)][col]
// =====================================================================
__global__ void pack_weights_kernel(const float* __restrict__ W,
                                    const float* __restrict__ Uw,
                                    __bf16* __restrict__ Wpk) {
  int idx = blockIdx.x * blockDim.x + threadIdx.x;   // over 64*40*32*16
  if (idx >= 64 * KT_N * 32 * 16) return;
  int j    = idx & 15;
  int lane = (idx >> 4) & 31;
  int kt   = (idx >> 9) % KT_N;
  int tn   = idx / (KT_N * 512);
  int col  = tn * 16 + (lane & 15);
  int k    = kt * 32 + ((lane >> 4) * 16) + j;
  float v;
  if (k < 256)       v = W[(size_t)k * N4 + col];
  else if (k < 512)  v = Uw[(size_t)(k - 256) * N4 + col];
  else               v = W[(size_t)(256 + (k - 512)) * N4 + col];
  Wpk[idx] = (__bf16)v;
}

// Prep 2: pack Wo [256 x 48] into bf16 B-operand order: [tn(3)][kt(8)][lane(32)][16]
__global__ void pack_wo_kernel(const float* __restrict__ Wo,
                               __bf16* __restrict__ WoPk) {
  int idx = blockIdx.x * blockDim.x + threadIdx.x;   // over 3*8*32*16 = 12288
  if (idx >= 3 * 8 * 32 * 16) return;
  int j    = idx & 15;
  int lane = (idx >> 4) & 31;
  int kt   = (idx >> 9) % 8;
  int tn   = idx / (8 * 512);
  int col  = tn * 16 + (lane & 15);                  // < 48 always
  int k    = kt * 32 + ((lane >> 4) * 16) + j;       // < 256
  WoPk[idx] = (__bf16)Wo[(size_t)k * VV + col];
}

// Prep 3: convert encoder inputs to bf16 (TDM source).
__global__ void cvt_inputs_kernel(const float* __restrict__ in,
                                  __bf16* __restrict__ out, int n) {
  int i = (blockIdx.x * blockDim.x + threadIdx.x) * 4;
  if (i + 3 < n) {
    out[i + 0] = (__bf16)in[i + 0];
    out[i + 1] = (__bf16)in[i + 1];
    out[i + 2] = (__bf16)in[i + 2];
    out[i + 3] = (__bf16)in[i + 3];
  } else {
    for (int j = i; j < n; ++j) out[j] = (__bf16)in[j];
  }
}

// Branchless activations (saturate correctly at +/-inf via __expf).
__device__ __forceinline__ float sigmoidf_(float x) {
  return 1.0f / (1.0f + __expf(-x));
}
__device__ __forceinline__ float tanhf_(float x) {
  return 1.0f - 2.0f / (__expf(2.0f * x) + 1.0f);
}

// Issue one 1-D TDM tile load: 768 bf16 from global -> LDS.
__device__ __forceinline__ void tdm_row_load(unsigned lds_addr, uint64_t gaddr) {
  u32x4 g0;
  g0[0] = 1u;                                            // count=1, no gather
  g0[1] = lds_addr;                                      // LDS byte address
  g0[2] = (unsigned)(gaddr & 0xFFFFFFFFu);               // global addr lo
  g0[3] = (unsigned)((gaddr >> 32) & 0x01FFFFFFu) | (2u << 30);  // addr hi | type=2
  i32x8 g1;
  g1[0] = 0x00010000;          // data_size=1 (2 bytes); wg_mask=0; no pad/iterate
  g1[1] = (int)(DENC << 16);   // tensor_dim0 = 768 (lo16 in [31:16])
  g1[2] = 0x00010000;          // tensor_dim1 = 1
  g1[3] = (int)(DENC << 16);   // tile_dim0 = 768
  g1[4] = 0;                   // tile_dim1 = 0 (1-D), tile_dim2 = 0
  g1[5] = DENC;                // tensor_dim0_stride = 768
  g1[6] = 0;
  g1[7] = 0;
  i32x4 gz = {0, 0, 0, 0};
#if defined(__clang_major__) && (__clang_major__ >= 23)
  i32x8 gz8 = {0, 0, 0, 0, 0, 0, 0, 0};
  __builtin_amdgcn_tensor_load_to_lds(g0, g1, gz, gz, gz8, 0);
#else
  __builtin_amdgcn_tensor_load_to_lds(g0, g1, gz, gz, 0);
#endif
}

__device__ __forceinline__ v16bf cat16(v8bf a0, v8bf a1) {
  return __builtin_shufflevector(a0, a1,
           0,1,2,3,4,5,6,7,8,9,10,11,12,13,14,15);
}

// =====================================================================
// Persistent decoder: 8 workgroups x 512 threads (16 waves).
// Each workgroup owns 16 batch rows through all S steps.
// Wave w computes N-tiles {w, w+16, w+32, w+48} = gates i/f/g/o of
// units [16w, 16w+16)  -> LSTM elementwise fully in registers.
// Logits: waves 0..2 do a second WMMA pass on the bf16 h in LDS.
// =====================================================================
__global__ void __launch_bounds__(512)
decode_kernel(const int*   __restrict__ word_ids,
              const float* __restrict__ embed,
              const float* __restrict__ bvec,
              const float* __restrict__ bo_g,
              const __bf16* __restrict__ Wpk,
              const __bf16* __restrict__ WoPk,
              const __bf16* __restrict__ inp_bf,
              float* __restrict__ out,
              int S, int maxlab) {
  extern __shared__ __align__(16) char smem[];
  __bf16* xlds   = (__bf16*)smem;                              // [16][XSTR]  emb|h|hard
  float*  logits = (float*)(smem + ROWS * XSTR * 2);           // [16][48]
  float*  bestv  = logits + ROWS * VV;                         // [16][4] argmax partials
  int*    st     = (int*)(bestv + ROWS * 4);
  int* besti   = st;            // [16*4]
  int* targets = st + 64;       // [16]
  int* hai     = st + 80;       // [16]
  int* reps    = st + 96;       // [16]
  int* tsteps  = st + 112;      // [16]
  int* adj     = st + 128;      // [16]

  const int tid     = threadIdx.x;
  const int wave    = tid >> 5;
  const int lane    = tid & 31;
  const int rowbase = blockIdx.x * ROWS;
  const int unit    = wave * 16 + (lane & 15);   // this lane's LSTM unit
  const int hilo    = lane >> 4;                 // 0: rows 0..7, 1: rows 8..15

  // ---- one-time init ----
  if (tid < ROWS) {
    int cnt = 0;
    for (int j = 0; j < T_; ++j) cnt += (word_ids[(rowbase + tid) * T_ + j] != 0);
    tsteps[tid]  = cnt;
    targets[tid] = 1;   // BOS
    hai[tid]     = 0;
    reps[tid]    = 0;
  }
  for (int i = tid; i < ROWS * UU; i += 512) {   // zero h section of x
    int r = i >> 8, u = i & 255;
    xlds[r * XSTR + 256 + u] = (__bf16)0.0f;
  }
  __syncthreads();

  // per-lane gate biases (gate order i,f,g,o)
  const float b_i = bvec[0 * UU + unit];
  const float b_f = bvec[1 * UU + unit];
  const float b_g = bvec[2 * UU + unit];
  const float b_o = bvec[3 * UU + unit];
  // per-lane output bias (logits pass, waves 0..2 only)
  const float bo_reg = (wave < 3) ? bo_g[wave * 16 + (lane & 15)] : 0.0f;

  float creg[8];
#pragma unroll
  for (int r = 0; r < 8; ++r) creg[r] = 0.0f;

  const int m = lane & 15;
  const __bf16* xrow = xlds + (size_t)m * XSTR;      // A-operand row for this lane
  const __bf16* xh   = xrow + 256;                   // h section (K=256 for logits)
  const int aoff = hilo * 16;

  for (int s = 0; s < S; ++s) {
    // Opaque zero offset (SGPR) regenerated each step: weight-load addresses
    // depend on it, so LICM can never hoist the weight stream out of the step
    // loop (prevents the spill-to-scratch failure mode), while the base
    // pointers keep their kernarg provenance -> saddr-form global_load_b128.
    size_t blur = 0;
    asm volatile("" : "+s"(blur));
    const __bf16* wl  = Wpk + (size_t)lane * 16 + blur;
    const __bf16* pB0 = wl + (size_t)(wave +  0) * KT_N * 512;
    const __bf16* pB1 = wl + (size_t)(wave + 16) * KT_N * 512;
    const __bf16* pB2 = wl + (size_t)(wave + 32) * KT_N * 512;
    const __bf16* pB3 = wl + (size_t)(wave + 48) * KT_N * 512;
    const __bf16* pWo = WoPk + (size_t)wave * 8 * 512 + (size_t)lane * 16 + blur;

    // ---- 1) hard-attention index (computed by wave 0's low lanes) ----
    if (tid < ROWS) {
      int h = hai[tid];
      adj[tid] = (h < tsteps[tid]) ? h : h - 1;
    }
    // ---- 2) embedding fill: 512 threads x 8 elems ----
    {
      int r  = tid >> 5;
      int k0 = (tid & 31) * 8;
      const float* e = embed + (size_t)targets[r] * EE + k0;
#pragma unroll
      for (int j = 0; j < 8; ++j) xlds[r * XSTR + k0 + j] = (__bf16)e[j];
    }
    // ---- 3) TDM: 16 hard rows (bf16) straight into x's hard section ----
    if (wave == 0) {
      for (int r = 0; r < ROWS; ++r) {
        unsigned lds_addr = (unsigned)(uintptr_t)(&xlds[r * XSTR + 512]);
        uint64_t ga = (uint64_t)(uintptr_t)(inp_bf +
            ((size_t)(rowbase + r) * T_ + (size_t)adj[r]) * DENC);
        tdm_row_load(lds_addr, ga);
      }
      __builtin_amdgcn_s_wait_tensorcnt(0);
    }
    __syncthreads();

    // ---- 4) main GEMM: z = x[16x1280] @ Wcat (4 gate tiles per wave) ----
    v8f acc0 = {0.f,0.f,0.f,0.f,0.f,0.f,0.f,0.f};
    v8f acc1 = acc0, acc2 = acc0, acc3 = acc0;
#pragma unroll 4
    for (int kt = 0; kt < KT_N; ++kt) {
      const int kb = kt * 32;
      v8bf a0 = *(const v8bf*)(xrow + kb + aoff);
      v8bf a1 = *(const v8bf*)(xrow + kb + 8 + aoff);
      v16bf A  = cat16(a0, a1);
      v16bf B0 = *(const v16bf*)(pB0 + (size_t)kt * 512);
      v16bf B1 = *(const v16bf*)(pB1 + (size_t)kt * 512);
      v16bf B2 = *(const v16bf*)(pB2 + (size_t)kt * 512);
      v16bf B3 = *(const v16bf*)(pB3 + (size_t)kt * 512);
      acc0 = __builtin_amdgcn_wmma_f32_16x16x32_bf16(false, A, false, B0, (short)0, acc0, false, false);
      acc1 = __builtin_amdgcn_wmma_f32_16x16x32_bf16(false, A, false, B1, (short)0, acc1, false, false);
      acc2 = __builtin_amdgcn_wmma_f32_16x16x32_bf16(false, A, false, B2, (short)0, acc2, false, false);
      acc3 = __builtin_amdgcn_wmma_f32_16x16x32_bf16(false, A, false, B3, (short)0, acc3, false, false);
    }

    // ---- 5) LSTM elementwise, fully in registers (branchless activations) ----
    float hnew[8];
#pragma unroll
    for (int r = 0; r < 8; ++r) {
      float zi = acc0[r] + b_i;
      float zf = acc1[r] + b_f;
      float zg = acc2[r] + b_g;
      float zo = acc3[r] + b_o;
      float cn = sigmoidf_(zf) * creg[r] + sigmoidf_(zi) * tanhf_(zg);
      float hn = sigmoidf_(zo) * tanhf_(cn);
      creg[r] = cn;
      hnew[r] = hn;
    }
    __syncthreads();   // all waves done reading x for this step
#pragma unroll
    for (int r = 0; r < 8; ++r) {
      int row = r + (hilo << 3);
      xlds[row * XSTR + 256 + unit] = (__bf16)hnew[r];   // bf16 h: next GEMM + logits
    }
    __syncthreads();

    // ---- 6) logits = h @ Wo + bo via WMMA (waves 0..2, one N-tile each) ----
    if (wave < 3) {
      v8f la = {0.f,0.f,0.f,0.f,0.f,0.f,0.f,0.f};
#pragma unroll
      for (int kt = 0; kt < 8; ++kt) {
        const int kb = kt * 32;
        v8bf q0 = *(const v8bf*)(xh + kb + aoff);
        v8bf q1 = *(const v8bf*)(xh + kb + 8 + aoff);
        v16bf A  = cat16(q0, q1);
        v16bf Bv = *(const v16bf*)(pWo + (size_t)kt * 512);
        la = __builtin_amdgcn_wmma_f32_16x16x32_bf16(false, A, false, Bv, (short)0, la, false, false);
      }
      const int col = wave * 16 + (lane & 15);
#pragma unroll
      for (int r = 0; r < 8; ++r) {
        int row = r + (hilo << 3);
        logits[row * VV + col] = la[r] + bo_reg;
      }
    }
    __syncthreads();

    // ---- 7a) argmax partials: 64 threads = 16 rows x 4 segments of 12 ----
    if (tid < 64) {
      int r   = tid >> 2;
      int seg = tid & 3;
      const float* lr = logits + r * VV + seg * 12;
      float bv = lr[0];
      int   bi2 = 0;
      for (int v = 1; v < 12; ++v) {
        float l = lr[v];
        if (l > bv) { bv = l; bi2 = v; }
      }
      bestv[r * 4 + seg] = bv;
      besti[r * 4 + seg] = seg * 12 + bi2;
    }
    __syncthreads();

    // ---- 7b) combine + state machine (first-max-wins tie order) ----
    if (tid < ROWS) {
      int r = tid;
      float bv = bestv[r * 4];
      int   pred = besti[r * 4];
      for (int seg = 1; seg < 4; ++seg) {
        float v = bestv[r * 4 + seg];
        if (v > bv) { bv = v; pred = besti[r * 4 + seg]; }
      }
      if (reps[r] >= maxlab) pred = 2;                        // force EOW
      float res = (hai[r] == tsteps[r]) ? 0.0f : (float)pred; // BATCH_PAD
      out[(size_t)(rowbase + r) * S + s] = res;
      int inc = (pred == 2 && hai[r] < tsteps[r]) ? 1 : 0;
      hai[r] += inc;
      if (inc) reps[r] = 0;
      targets[r] = pred;
    }
    __syncthreads();
  }
}

// =====================================================================
extern "C" void kernel_launch(void* const* d_in, const int* in_sizes, int n_in,
                              void* d_out, int out_size, void* d_ws, size_t ws_size,
                              hipStream_t stream) {
  const float* inputs   = (const float*)d_in[0];
  const int*   word_ids = (const int*)d_in[1];
  const float* embed    = (const float*)d_in[2];
  const float* W        = (const float*)d_in[3];
  const float* Uw       = (const float*)d_in[4];
  const float* bvec     = (const float*)d_in[5];
  const float* Wo       = (const float*)d_in[6];
  const float* bo       = (const float*)d_in[7];
  float* out = (float*)d_out;

  const int S      = out_size / B_;        // 192
  const int maxlab = (S / T_ > 0) ? S / T_ : 1;

  // workspace layout
  const size_t wpk_elems  = (size_t)64 * KT_N * 32 * 16;   // 1,310,720 bf16
  const size_t wopk_elems = (size_t)3 * 8 * 32 * 16;       // 12,288 bf16
  const int    ninp       = B_ * T_ * DENC;
  size_t off = 0;
  __bf16* Wpk    = (__bf16*)((char*)d_ws + off); off += (wpk_elems * 2 + 255) & ~(size_t)255;
  __bf16* WoPk   = (__bf16*)((char*)d_ws + off); off += (wopk_elems * 2 + 255) & ~(size_t)255;
  __bf16* inp_bf = (__bf16*)((char*)d_ws + off);

  pack_weights_kernel<<<(int)((wpk_elems + 255) / 256), 256, 0, stream>>>(W, Uw, Wpk);
  pack_wo_kernel<<<(int)((wopk_elems + 255) / 256), 256, 0, stream>>>(Wo, WoPk);
  cvt_inputs_kernel<<<(ninp / 4 + 255) / 256, 256, 0, stream>>>(inputs, inp_bf, ninp);

  const size_t smem = (size_t)ROWS * XSTR * 2      // x  (bf16)
                    + (size_t)ROWS * VV * 4        // logits
                    + (size_t)ROWS * 4 * 4         // bestv
                    + (size_t)ROWS * 4 * 4         // besti
                    + 5 * ROWS * 4;                // state
  decode_kernel<<<B_ / ROWS, 512, smem, stream>>>(
      word_ids, embed, bvec, bo, Wpk, WoPk, inp_bf, out, S, maxlab);
}
